// GCN_adaboost_content_35871566856589
// MI455X (gfx1250) — matrix operand order repaired and checked by reading, not compiled, not run on hardware
//
#include <hip/hip_runtime.h>
#include <cstdint>
#include <cstddef>

// ---------------------------------------------------------------------------
// Types for CDNA5 WMMA (wave32): V_WMMA_F32_16X16X32_BF16
// ---------------------------------------------------------------------------
typedef __bf16 bf16;
typedef __attribute__((ext_vector_type(16))) __bf16 v16bf;
typedef __attribute__((ext_vector_type(8)))  __bf16 bf16x8;
typedef __attribute__((ext_vector_type(8)))  float  v8f;

union Frag16 { v16bf v; bf16x8 h[2]; };

#define N_NODES 50000
#define N_EDGES 800000
#define F_IN    512
#define H_MID   128
#define H_OUT   64
#define N_CLS   7
#define M_IDX   10000

#define SLAB_K  128     // K-slab staged into LDS per iteration
#define PAD_K   8       // bf16 row padding -> 16B; makes ds_load_b128 bank-conflict-free

// ---------------------------------------------------------------------------
// Conversions
// ---------------------------------------------------------------------------
__global__ void k_f32_to_bf16(const float* __restrict__ in, bf16* __restrict__ out,
                              long long n) {
  long long i = (long long)blockIdx.x * blockDim.x + threadIdx.x;
  if (i < n) out[i] = (bf16)in[i];
}

__global__ void k_relu_to_bf16(const float* __restrict__ in, bf16* __restrict__ out,
                               long long n) {
  long long i = (long long)blockIdx.x * blockDim.x + threadIdx.x;
  if (i < n) {
    float v = in[i];
    out[i] = (bf16)(v > 0.f ? v : 0.f);
  }
}

// Wt[n][k] = (bf16) W[k][n]   (W is K x Kout row-major)
__global__ void k_w_transpose_bf16(const float* __restrict__ W, bf16* __restrict__ Wt,
                                   int K, int kout) {
  int i = blockIdx.x * blockDim.x + threadIdx.x;
  if (i < K * kout) {
    int k = i / kout, n = i - k * kout;
    Wt[(size_t)n * K + k] = (bf16)W[i];
  }
}

// ---------------------------------------------------------------------------
// BF16 WMMA GEMM with LDS-staged weights:
//   C[nrows x kout] = A[nrows x K](bf16) * Bt[kout x K](bf16)^T
// One wave owns a 16-row x kout strip (COLB = kout/16 tiles), reusing its A
// fragment across all column tiles. The 256-thread workgroup stages each
// kout x SLAB_K weight slab into LDS once and 8 waves share it.
// Per k-step: preload all COLB B-fragments, then run the WMMA chain
// uninterrupted (single s_wait_dscnt, back-to-back matrix ops).
// K % SLAB_K == 0, nrows % 16 == 0.
// ---------------------------------------------------------------------------
template <int COLB>
__global__ void k_gemm_bf16_lds(const bf16* __restrict__ A, const bf16* __restrict__ Bt,
                                float* __restrict__ C, int nrows, int K) {
  constexpr int ROWS_B = COLB * 16;          // weight rows (= kout)
  constexpr int LROW   = SLAB_K + PAD_K;     // padded LDS row length (bf16)
  __shared__ bf16 lds[ROWS_B * LROW];

  const int kout  = COLB * 16;
  const int tid   = threadIdx.x;
  const int gwave = (int)(((long long)blockIdx.x * blockDim.x + tid) >> 5);
  const int lane  = tid & 31;
  const int row0  = gwave << 4;
  const bool active = (row0 < nrows);        // uniform per wave
  const int half = lane >> 4;
  const int l    = lane & 15;

  const bf16* arow = A + (size_t)(row0 + l) * K;  // A lane holds row M=l

  v8f acc[COLB];
#pragma unroll
  for (int j = 0; j < COLB; ++j) acc[j] = (v8f){};

  for (int kk0 = 0; kk0 < K; kk0 += SLAB_K) {
    __syncthreads();
    // ---- cooperative stage: Bt[0..kout)[kk0..kk0+SLAB_K) -> LDS (b128 copies)
    constexpr int CHUNKS = ROWS_B * (SLAB_K / 8);   // 8 bf16 per 16B chunk
    for (int c = tid; c < CHUNKS; c += 256) {
      int n    = c >> 4;              // SLAB_K/8 == 16 chunks per row
      int koff = (c & 15) << 3;
      *(bf16x8*)(lds + n * LROW + koff) =
          *(const bf16x8*)(Bt + (size_t)n * K + kk0 + koff);
    }
    __syncthreads();

    if (active) {
      for (int kk = 0; kk < SLAB_K; kk += 32) {
        // 16-bit A 16x32 layout: lane<16 -> K 0..7 & 16..23 ; lane>=16 -> 8..15 & 24..31
        Frag16 a;
        a.h[0] = *(const bf16x8*)(arow + kk0 + kk + half * 8);
        a.h[1] = *(const bf16x8*)(arow + kk0 + kk + half * 8 + 16);
        // Preload all B fragments for this k-step...
        Frag16 b[COLB];
#pragma unroll
        for (int j = 0; j < COLB; ++j) {
          // 16-bit B 32x16 layout: lane<16 -> K 0..7 & 16..23 ; lane>=16 -> 8..15 & 24..31
          const bf16* brow = lds + (j * 16 + l) * LROW;
          b[j].h[0] = *(const bf16x8*)(brow + kk + half * 8);
          b[j].h[1] = *(const bf16x8*)(brow + kk + half * 8 + 16);
        }
        // ...then issue the matrix-op chain back-to-back.
#pragma unroll
        for (int j = 0; j < COLB; ++j)
          acc[j] = __builtin_amdgcn_wmma_f32_16x16x32_bf16(false, a.v, false, b[j].v,
                                                           (short)0, acc[j], false, false);
      }
    }
  }

  if (active) {
    // C/D layout: VGPR r -> M = half*8 + r, N = l
#pragma unroll
    for (int j = 0; j < COLB; ++j) {
      float* cptr = C + (size_t)(row0 + half * 8) * kout + j * 16 + l;
#pragma unroll
      for (int r = 0; r < 8; ++r) cptr[(size_t)r * kout] = acc[j][r];
    }
  }
}

// ---------------------------------------------------------------------------
// SPMM pieces: out[row] += w[e] * src[col]  (bias pre-broadcast, f32 atomics)
// ---------------------------------------------------------------------------
__global__ void k_bias_init(float* __restrict__ out, const float* __restrict__ b,
                            long long n, int mask) {
  long long i = (long long)blockIdx.x * blockDim.x + threadIdx.x;
  if (i < n) out[i] = b[(int)(i & mask)];
}

__global__ void k_spmm_scatter(const int* __restrict__ idx, const float* __restrict__ w,
                               const float* __restrict__ src, float* __restrict__ dst,
                               int E, int dimShift) {
  long long t = (long long)blockIdx.x * blockDim.x + threadIdx.x;
  int dim = 1 << dimShift;
  long long e = t >> dimShift;
  int f = (int)(t & (dim - 1));
  if (e >= E) return;
  int row = idx[e];                 // idx[0][e]: segment (destination) row
  int col = idx[E + e];             // idx[1][e]: gather row
  float v = w[e] * src[(size_t)col * dim + f];
  atomicAdd(dst + (size_t)row * dim + f, v);
}

// ---------------------------------------------------------------------------
// Dense heads
// ---------------------------------------------------------------------------
// out[n][c] = sum_k relu(X[n][k]) * W[k][c] + b[c]   (kin=64, kout=7)
__global__ void k_dense_relu(const float* __restrict__ X, const float* __restrict__ W,
                             const float* __restrict__ b, float* __restrict__ out,
                             int n, int kin, int kout) {
  int i = blockIdx.x * blockDim.x + threadIdx.x;
  if (i >= n * kout) return;
  int r = i / kout, c = i - r * kout;
  const float* xr = X + (size_t)r * kin;
  float s = b[c];
  for (int k = 0; k < kin; ++k) {
    float v = xr[k];
    v = v > 0.f ? v : 0.f;
    s += v * W[k * kout + c];
  }
  out[i] = s;
}

// sim_dense = concat(x2,x3,x6) @ simW + simb  (raw, no relu; 3*64 -> 7)
__global__ void k_sim_dense(const float* __restrict__ X2, const float* __restrict__ X3,
                            const float* __restrict__ X6, const float* __restrict__ W,
                            const float* __restrict__ b, float* __restrict__ out, int n) {
  int i = blockIdx.x * blockDim.x + threadIdx.x;
  if (i >= n * N_CLS) return;
  int r = i / N_CLS, c = i - r * N_CLS;
  const float* x2 = X2 + (size_t)r * H_OUT;
  const float* x3 = X3 + (size_t)r * H_OUT;
  const float* x6 = X6 + (size_t)r * H_OUT;
  float s = b[c];
  for (int k = 0; k < H_OUT; ++k) {
    s += x2[k] * W[k * N_CLS + c];
    s += x3[k] * W[(H_OUT + k) * N_CLS + c];
    s += x6[k] * W[(2 * H_OUT + k) * N_CLS + c];
  }
  out[i] = s;
}

// ---------------------------------------------------------------------------
// AdaBoost reductions
// ---------------------------------------------------------------------------
__global__ void k_zero(float* p, int n) {
  int i = blockIdx.x * blockDim.x + threadIdx.x;
  if (i < n) p[i] = 0.f;
}

// t = exp(-(a[idx]*yi)); sumT += t;  if (g[idx]*yi >= 0) sumG += t;
__global__ void k_boost_reduce(const float* __restrict__ a, const float* __restrict__ g,
                               const float* __restrict__ y, const int* __restrict__ index,
                               float* sumT, float* sumG, int M) {
  int i = blockIdx.x * blockDim.x + threadIdx.x;
  if (i >= M * N_CLS) return;
  int m = i / N_CLS, c = i - m * N_CLS;
  int n = index[m];
  size_t o = (size_t)n * N_CLS + c;
  float yi = y[o];
  float t = __expf(-(a[o] * yi));
  atomicAdd(sumT, t);
  if (g[o] * yi >= 0.f) atomicAdd(sumG, t);
}

// alpha = 0.5*log(sumG/(sumT-sumG)); out = base + add*alpha
__global__ void k_axpy_alpha(const float* __restrict__ base, const float* __restrict__ add,
                             const float* __restrict__ sumT, const float* __restrict__ sumG,
                             float* __restrict__ out, long long n) {
  long long i = (long long)blockIdx.x * blockDim.x + threadIdx.x;
  if (i >= n) return;
  float sG = *sumG;
  float alpha = 0.5f * __logf(sG / (*sumT - sG));
  out[i] = base[i] + add[i] * alpha;
}

// ---------------------------------------------------------------------------
static inline int nblk(long long n, int bs) { return (int)((n + bs - 1) / bs); }

extern "C" void kernel_launch(void* const* d_in, const int* in_sizes, int n_in,
                              void* d_out, int out_size, void* d_ws, size_t ws_size,
                              hipStream_t stream) {
  (void)in_sizes; (void)n_in; (void)out_size; (void)ws_size;
  const float* x = (const float*)d_in[0];
  const int*   adj_idx[7];
  const float* adj_w[7];
  for (int i = 1; i <= 6; ++i) {
    adj_idx[i] = (const int*)  d_in[1 + 2 * (i - 1)];
    adj_w[i]   = (const float*)d_in[2 + 2 * (i - 1)];
  }
  const float* y     = (const float*)d_in[13];
  const int*   index = (const int*)  d_in[14];
  const float* W_[9]; const float* B_[9];                 // gc1..gc6, gc10..gc12
  for (int i = 0; i < 9; ++i) {
    W_[i] = (const float*)d_in[15 + 2 * i];
    B_[i] = (const float*)d_in[16 + 2 * i];
  }
  const float* d1W = (const float*)d_in[33], *d1b = (const float*)d_in[34];
  const float* d2W = (const float*)d_in[35], *d2b = (const float*)d_in[36];
  const float* d3W = (const float*)d_in[37], *d3b = (const float*)d_in[38];
  const float* d4W = (const float*)d_in[39], *d4b = (const float*)d_in[40];
  const float* simW = (const float*)d_in[41], *simb = (const float*)d_in[42];

  // ---- workspace carve ----
  char* ws = (char*)d_ws;
  size_t off = 0;
  auto alloc = [&](size_t bytes) -> void* {
    off = (off + 255) & ~(size_t)255;
    void* p = ws + off;
    off += bytes;
    return p;
  };
  bf16*  xb  = (bf16*) alloc((size_t)N_NODES * F_IN * sizeof(bf16));
  float* hw  = (float*)alloc((size_t)N_NODES * H_MID * sizeof(float));
  float* h   = (float*)alloc((size_t)N_NODES * H_MID * sizeof(float));
  bf16*  hb  = (bf16*) alloc((size_t)N_NODES * H_MID * sizeof(bf16));
  float* bx1 = (float*)alloc((size_t)N_NODES * H_OUT * sizeof(float));
  float* bx2 = (float*)alloc((size_t)N_NODES * H_OUT * sizeof(float));
  float* bx3 = (float*)alloc((size_t)N_NODES * H_OUT * sizeof(float));
  float* bx6 = (float*)alloc((size_t)N_NODES * H_OUT * sizeof(float));
  float* bx4 = (float*)alloc((size_t)N_NODES * H_OUT * sizeof(float));
  static const int Ks[3]  = {F_IN, H_MID, H_MID};
  static const int Kos[3] = {H_MID, H_MID, H_OUT};
  bf16* wt[9];
  for (int i = 0; i < 9; ++i)
    wt[i] = (bf16*)alloc((size_t)Ks[i % 3] * Kos[i % 3] * sizeof(bf16));
  float* x1d   = (float*)alloc((size_t)N_NODES * N_CLS * sizeof(float));
  float* x4d   = (float*)alloc((size_t)N_NODES * N_CLS * sizeof(float));
  float* simd_ = (float*)alloc((size_t)N_NODES * N_CLS * sizeof(float));
  float* part1 = (float*)alloc((size_t)N_NODES * N_CLS * sizeof(float));
  float* sums  = (float*)alloc(16 * sizeof(float));

  const int BS = 256;

  // ---- one-time converts ----
  { long long n = (long long)N_NODES * F_IN;
    k_f32_to_bf16<<<nblk(n, BS), BS, 0, stream>>>(x, xb, n); }
  for (int i = 0; i < 9; ++i) {
    int n = Ks[i % 3] * Kos[i % 3];
    k_w_transpose_bf16<<<nblk(n, BS), BS, 0, stream>>>(W_[i], wt[i], Ks[i % 3], Kos[i % 3]);
  }

  auto gemm = [&](const bf16* A, const bf16* Bt, float* C, int K, int kout) {
    long long waves   = N_NODES / 16;          // one wave per 16-row strip
    long long threads = waves * 32;
    int blocks = nblk(threads, BS);
    if (kout == 128)
      k_gemm_bf16_lds<8><<<blocks, BS, 0, stream>>>(A, Bt, C, N_NODES, K);
    else
      k_gemm_bf16_lds<4><<<blocks, BS, 0, stream>>>(A, Bt, C, N_NODES, K);
  };
  auto spmm = [&](const int* ai, const float* aw, const float* src, float* dst,
                  const float* bias, int dim, int shift) {
    long long nb = (long long)N_NODES * dim;
    k_bias_init<<<nblk(nb, BS), BS, 0, stream>>>(dst, bias, nb, dim - 1);
    long long ne = (long long)N_EDGES * dim;
    k_spmm_scatter<<<nblk(ne, BS), BS, 0, stream>>>(ai, aw, src, dst, N_EDGES, shift);
  };
  auto run_branch = [&](const int* ai, const float* aw, int wb, float* outB) {
    gemm(xb, wt[wb + 0], hw, F_IN, H_MID);
    spmm(ai, aw, hw, h, B_[wb + 0], H_MID, 7);
    { long long n = (long long)N_NODES * H_MID;
      k_relu_to_bf16<<<nblk(n, BS), BS, 0, stream>>>(h, hb, n); }
    gemm(hb, wt[wb + 1], hw, H_MID, H_MID);
    spmm(ai, aw, hw, h, B_[wb + 1], H_MID, 7);
    { long long n = (long long)N_NODES * H_MID;
      k_relu_to_bf16<<<nblk(n, BS), BS, 0, stream>>>(h, hb, n); }
    gemm(hb, wt[wb + 2], hw, H_MID, H_OUT);
    spmm(ai, aw, hw, outB, B_[wb + 2], H_OUT, 6);   // no relu on branch output
  };

  // ---- five branches ----
  run_branch(adj_idx[5], adj_w[5], 0, bx1);   // x1: adj5, gc1/gc2/gc3
  run_branch(adj_idx[4], adj_w[4], 3, bx2);   // x2: adj4, gc4/gc5/gc6
  run_branch(adj_idx[3], adj_w[3], 3, bx3);   // x3: adj3, gc4/gc5/gc6
  run_branch(adj_idx[6], adj_w[6], 3, bx6);   // x6: adj6, gc4/gc5/gc6
  run_branch(adj_idx[1], adj_w[1], 6, bx4);   // x4: adj1, gc10/gc11/gc12

  // ---- dense heads (x2/x3/x6 heads go straight to d_out) ----
  float* out_x2d = (float*)d_out;
  float* out_x3d = out_x2d + (size_t)N_NODES * N_CLS;
  float* out_x6d = out_x3d + (size_t)N_NODES * N_CLS;
  float* out_p2  = out_x6d + (size_t)N_NODES * N_CLS;
  long long nd = (long long)N_NODES * N_CLS;
  k_dense_relu<<<nblk(nd, BS), BS, 0, stream>>>(bx1, d1W, d1b, x1d,     N_NODES, H_OUT, N_CLS);
  k_dense_relu<<<nblk(nd, BS), BS, 0, stream>>>(bx2, d2W, d2b, out_x2d, N_NODES, H_OUT, N_CLS);
  k_dense_relu<<<nblk(nd, BS), BS, 0, stream>>>(bx3, d3W, d3b, out_x3d, N_NODES, H_OUT, N_CLS);
  k_dense_relu<<<nblk(nd, BS), BS, 0, stream>>>(bx6, d3W, d3b, out_x6d, N_NODES, H_OUT, N_CLS); // torch reuses dense3
  k_dense_relu<<<nblk(nd, BS), BS, 0, stream>>>(bx4, d4W, d4b, x4d,     N_NODES, H_OUT, N_CLS);
  k_sim_dense <<<nblk(nd, BS), BS, 0, stream>>>(bx2, bx3, bx6, simW, simb, simd_, N_NODES);

  // ---- adaboost alphas + outputs ----
  k_zero<<<1, 16, 0, stream>>>(sums, 16);
  long long nr = (long long)M_IDX * N_CLS;
  k_boost_reduce<<<nblk(nr, BS), BS, 0, stream>>>(x4d, simd_, y, index, &sums[0], &sums[1], M_IDX);
  k_axpy_alpha  <<<nblk(nd, BS), BS, 0, stream>>>(x4d, simd_, &sums[0], &sums[1], part1, nd);
  k_boost_reduce<<<nblk(nr, BS), BS, 0, stream>>>(part1, x1d, y, index, &sums[2], &sums[3], M_IDX);
  k_axpy_alpha  <<<nblk(nd, BS), BS, 0, stream>>>(part1, x1d, &sums[2], &sums[3], out_p2, nd);
}
